// NashSuruMoE_30030411334251
// MI455X (gfx1250) — compile-verified, matrix operand order checked by reading, not run on hardware
//
#include <hip/hip_runtime.h>
#include <hip/hip_bf16.h>
#include <math.h>

// ---------------- types ----------------
typedef __attribute__((ext_vector_type(16))) __bf16    v16bf;
typedef __attribute__((ext_vector_type(8)))  float     v8f;
typedef __attribute__((ext_vector_type(4))) unsigned   u32x4;
typedef __attribute__((ext_vector_type(2))) unsigned   u32x2;
typedef __attribute__((ext_vector_type(4)))  float     f32x4;

union ABfrag { v16bf v; u32x4 q[2]; };

// ---------------- problem constants ----------------
#define NB   2
#define NS   2048
#define ND   1024
#define NE   8
#define NH   4096
#define NTOK (NB*NS)      // 4096
#define CAP  (NTOK*2)     // worst-case tokens per expert list
#define MT   64           // token M-tile in FFN kernel
#define HC   64           // H chunk per iteration

__device__ __forceinline__ unsigned short f2bf(float f) {
    union { float f; unsigned u; } v; v.f = f;
    unsigned u = v.u;
    u += 0x7fffu + ((u >> 16) & 1u);          // round-to-nearest-even
    return (unsigned short)(u >> 16);
}

// ---------------- 1) fp32 -> bf16 tiled transpose (per expert slice) ------
// in:  (R x C) row-major fp32     out: (C x R) row-major bf16
__global__ __launch_bounds__(256) void transpose_bf16_kernel(
    const float* __restrict__ in, unsigned short* __restrict__ out, int R, int C)
{
    __shared__ unsigned short tileS[32][33];
    const size_t eoff = (size_t)blockIdx.z * (size_t)R * (size_t)C;
    in  += eoff;
    out += eoff;
    const int tx = threadIdx.x & 31;
    const int ty = threadIdx.x >> 5;          // 0..7
    const int c0 = blockIdx.x * 32;
    const int r0 = blockIdx.y * 32;
    #pragma unroll
    for (int r = 0; r < 32; r += 8) {
        const int rr = ty + r;
        float v = in[(size_t)(r0 + rr) * C + (c0 + tx)];
        tileS[tx][rr] = f2bf(v);              // tileS[c_local][r_local]
    }
    __syncthreads();
    #pragma unroll
    for (int r = 0; r < 32; r += 8) {
        const int rr = ty + r;
        out[(size_t)(c0 + rr) * R + (r0 + tx)] = tileS[rr][tx];
    }
}

// ---------------- 2) expert embedding squared norms ----------------
__global__ void enorm_kernel(const float* __restrict__ emb, float* __restrict__ enorm2)
{
    const int e = threadIdx.x >> 5;           // 8 waves -> 8 experts
    const int lane = threadIdx.x & 31;
    float s = 0.f;
    for (int i = lane; i < ND; i += 32) { float v = emb[e * ND + i]; s += v * v; }
    #pragma unroll
    for (int m = 16; m >= 1; m >>= 1) s += __shfl_xor(s, m, 32);
    if (lane == 0) enorm2[e] = s;
}

// ---------------- 3) routing: one wave32 per token ----------------
__global__ __launch_bounds__(256) void route_kernel(
    const float* __restrict__ x, const float* __restrict__ emb,
    const float* __restrict__ enorm2, const float* __restrict__ cache,
    int* __restrict__ eidx, float* __restrict__ ewgt)
{
    const int token = blockIdx.x * 8 + (threadIdx.x >> 5);
    const int lane  = threadIdx.x & 31;
    if (token >= NTOK) return;

    float t2 = 0.f;
    float dot[NE];
    #pragma unroll
    for (int e = 0; e < NE; ++e) dot[e] = 0.f;
    const float* xr = x + (size_t)token * ND;
    for (int i = lane; i < ND; i += 32) {
        const float xv = xr[i];
        t2 += xv * xv;
        #pragma unroll
        for (int e = 0; e < NE; ++e) dot[e] += xv * emb[e * ND + i];
    }
    #pragma unroll
    for (int m = 16; m >= 1; m >>= 1) {
        t2 += __shfl_xor(t2, m, 32);
        #pragma unroll
        for (int e = 0; e < NE; ++e) dot[e] += __shfl_xor(dot[e], m, 32);
    }
    if (lane != 0) return;

    // drop the lowest-sim expert (top-7 of 8); <= keeps the LAST min excluded
    int emin = 0; float smin = dot[0];
    #pragma unroll
    for (int e = 1; e < NE; ++e) if (dot[e] <= smin) { smin = dot[e]; emin = e; }

    const float tn = sqrtf(t2) + 1e-8f;
    float sc[NE];
    #pragma unroll
    for (int e = 0; e < NE; ++e)
        sc[e] = dot[e] / (tn * (sqrtf(enorm2[e]) + 1e-8f)) + 0.1f * cache[e];

    int b0 = -1, b1i = -1; float s0 = -1e30f, s1 = -1e30f;
    #pragma unroll
    for (int e = 0; e < NE; ++e) {
        if (e == emin) continue;
        const float s = sc[e];
        if (s > s0 || (s == s0 && (b0 < 0 || dot[e] > dot[b0]))) {
            s1 = s0; b1i = b0; s0 = s; b0 = e;
        } else if (s > s1 || (s == s1 && (b1i < 0 || dot[e] > dot[b1i]))) {
            s1 = s; b1i = e;
        }
    }
    const float mx = fmaxf(s0, s1);
    const float e0 = __expf(s0 - mx), e1 = __expf(s1 - mx);
    const float inv = 1.f / (e0 + e1);
    eidx[token * 2 + 0] = b0;  ewgt[token * 2 + 0] = e0 * inv;
    eidx[token * 2 + 1] = b1i; ewgt[token * 2 + 1] = e1 * inv;
}

// ---------------- 4) zero per-expert counters ----------------
__global__ void zero_counts_kernel(int* __restrict__ counts)
{
    if (threadIdx.x < NE) counts[threadIdx.x] = 0;
}

// ---------------- 5) scatter tokens into per-expert lists ----------------
__global__ __launch_bounds__(256) void scatter_kernel(
    const int* __restrict__ eidx, const float* __restrict__ ewgt,
    int* __restrict__ counts, int* __restrict__ ltok, float* __restrict__ lwt)
{
    const int n = blockIdx.x * blockDim.x + threadIdx.x;
    if (n >= NTOK) return;
    #pragma unroll
    for (int k = 0; k < 2; ++k) {
        const int e   = eidx[n * 2 + k];
        const float w = ewgt[n * 2 + k];
        const int slot = atomicAdd(&counts[e], 1);
        ltok[e * CAP + slot] = n;
        lwt [e * CAP + slot] = w;
    }
}

// ---------------- 6) init out with softmax-weighted b2 ----------------
__global__ __launch_bounds__(256) void init_out_kernel(
    const float* __restrict__ b2, const int* __restrict__ eidx,
    const float* __restrict__ ewgt, float* __restrict__ out)
{
    const int idx = blockIdx.x * blockDim.x + threadIdx.x;
    if (idx >= NTOK * ND) return;
    const int n = idx >> 10;          // / ND
    const int d = idx & (ND - 1);
    const int   e0 = eidx[n * 2 + 0], e1 = eidx[n * 2 + 1];
    const float w0 = ewgt[n * 2 + 0], w1 = ewgt[n * 2 + 1];
    out[idx] = w0 * b2[e0 * ND + d] + w1 * b2[e1 * ND + d];
}

// ---------------- 7) fused expert FFN: x@W1 -> +b1 -> gelu -> @W2 --------
// grid: (tile, expert); block: 256 threads = 8 wave32
// dyn LDS: X tile (MT x ND bf16) + h chunk (MT x HC bf16) + token list
__global__ __launch_bounds__(256) void ffn_kernel(
    const float* __restrict__ x,
    const unsigned short* __restrict__ w1t,   // (E,H,D) bf16
    const unsigned short* __restrict__ w2t,   // (E,D,H) bf16
    const float* __restrict__ b1,             // (E,H)
    const int* __restrict__ counts,
    const int* __restrict__ ltok, const float* __restrict__ lwt,
    float* __restrict__ out)
{
    const int e    = blockIdx.y;
    const int tile = blockIdx.x;
    const int cnt  = counts[e];
    if (tile * MT >= cnt) return;

    extern __shared__ char smem[];
    unsigned short* Xl   = (unsigned short*)smem;            // MT*ND
    unsigned short* Hl   = Xl + MT * ND;                     // MT*HC
    int*            TOKS = (int*)(Hl + MT * HC);             // MT
    float*          WGTS = (float*)(TOKS + MT);              // MT

    const int tid   = threadIdx.x;
    const int wid   = tid >> 5;
    const int lane  = tid & 31;
    const int l16   = lane & 15;
    const int khalf = lane >> 4;

    if (tid < MT) {
        const int r = tile * MT + tid;
        TOKS[tid] = (r < cnt) ? ltok[e * CAP + r] : -1;
        WGTS[tid] = (r < cnt) ? lwt [e * CAP + r] : 0.f;
    }
    __syncthreads();

    // stage X tile to LDS as bf16 (float4 reads, 8-byte LDS writes)
    for (int i = tid; i < MT * (ND / 4); i += 256) {
        const int row = i >> 8;                 // / (ND/4)
        const int c4  = (i & (ND / 4 - 1)) * 4;
        const int t   = TOKS[row];
        f32x4 v = (f32x4)0.f;
        if (t >= 0) v = *(const f32x4*)(x + (size_t)t * ND + c4);
        u32x2 p;
        p[0] = (unsigned)f2bf(v[0]) | ((unsigned)f2bf(v[1]) << 16);
        p[1] = (unsigned)f2bf(v[2]) | ((unsigned)f2bf(v[3]) << 16);
        *(u32x2*)(Xl + row * ND + c4) = p;
    }
    __syncthreads();

    // persistent output accumulators: 4 M-subtiles x 8 N-subtiles per wave
    v8f acc2[4][8];
    #pragma unroll
    for (int a = 0; a < 4; ++a)
        #pragma unroll
        for (int b = 0; b < 8; ++b) { v8f z = {}; acc2[a][b] = z; }

    // phase-A work split: same A fragment for both owned h-columns
    const int msubA = wid >> 1;               // 0..3
    const int nshb  = (wid & 1) * 2;          // 0 or 2

    #pragma unroll 1
    for (int hc = 0; hc < NH / HC; ++hc) {
        // ---- phase A: h chunk (MT x HC) = X @ W1 chunk, +b1, gelu ----
        {
            const int hg0 = hc * HC + nshb * 16 + l16;
            const size_t wb0 = ((size_t)e * NH + hg0) * ND + khalf * 8;
            const size_t wb1 = wb0 + (size_t)16 * ND;

            // warm next H-chunk of W1 (global_prefetch, L2-resident stream)
            if (hc + 1 < NH / HC) {
                __builtin_prefetch(w1t + wb0 + (size_t)HC * ND, 0, 1);
                __builtin_prefetch(w1t + wb1 + (size_t)HC * ND, 0, 1);
            }

            v8f acc0 = {}, acc1 = {};
            #pragma unroll 4
            for (int kb = 0; kb < ND / 32; ++kb) {
                ABfrag a;                      // shared A fragment (LDS)
                const int aoff = (msubA * 16 + l16) * ND + kb * 32 + khalf * 8;
                a.q[0] = *(const u32x4*)(Xl + aoff);
                a.q[1] = *(const u32x4*)(Xl + aoff + 16);
                ABfrag bf0, bf1;               // W1 fragments (base + imm offsets)
                bf0.q[0] = *(const u32x4*)(w1t + wb0 + kb * 32);
                bf0.q[1] = *(const u32x4*)(w1t + wb0 + kb * 32 + 16);
                bf1.q[0] = *(const u32x4*)(w1t + wb1 + kb * 32);
                bf1.q[1] = *(const u32x4*)(w1t + wb1 + kb * 32 + 16);
                acc0 = __builtin_amdgcn_wmma_f32_16x16x32_bf16(
                        false, a.v, false, bf0.v, (short)0, acc0, false, false);
                acc1 = __builtin_amdgcn_wmma_f32_16x16x32_bf16(
                        false, a.v, false, bf1.v, (short)0, acc1, false, false);
            }
            const float bias0 = b1[(size_t)e * NH + hg0];
            const float bias1 = b1[(size_t)e * NH + hg0 + 16];
            #pragma unroll
            for (int r = 0; r < 8; ++r) {
                const int m = r + 8 * khalf;
                float v0 = acc0[r] + bias0;
                v0 = 0.5f * v0 * (1.f + erff(v0 * 0.70710678118654752f));
                Hl[(msubA * 16 + m) * HC + nshb * 16 + l16] = f2bf(v0);
                float v1 = acc1[r] + bias1;
                v1 = 0.5f * v1 * (1.f + erff(v1 * 0.70710678118654752f));
                Hl[(msubA * 16 + m) * HC + (nshb + 1) * 16 + l16] = f2bf(v1);
            }
        }
        __syncthreads();

        // ---- phase B: out tile += h chunk @ W2 chunk ----
        if (hc + 1 < NH / HC)   // warm next H-chunk of W2
            __builtin_prefetch(
                w2t + ((size_t)e * ND + wid * 16 + l16) * NH + (hc + 1) * HC, 0, 1);

        #pragma unroll
        for (int kb2 = 0; kb2 < HC / 32; ++kb2) {
            ABfrag af[4];
            #pragma unroll
            for (int ms = 0; ms < 4; ++ms) {
                const int aoff = (ms * 16 + l16) * HC + kb2 * 32 + khalf * 8;
                af[ms].q[0] = *(const u32x4*)(Hl + aoff);
                af[ms].q[1] = *(const u32x4*)(Hl + aoff + 16);
            }
            // base pointer: j term is a pure 1MB stride -> 24-bit imm offsets
            const unsigned short* wbase =
                w2t + ((size_t)e * ND + wid * 16 + l16) * NH
                    + hc * HC + kb2 * 32 + khalf * 8;
            #pragma unroll
            for (int j = 0; j < 8; ++j) {
                ABfrag bf;
                const unsigned short* wp = wbase + (size_t)j * 8 * 16 * NH;
                bf.q[0] = *(const u32x4*)(wp);
                bf.q[1] = *(const u32x4*)(wp + 16);
                #pragma unroll
                for (int ms = 0; ms < 4; ++ms) {
                    acc2[ms][j] = __builtin_amdgcn_wmma_f32_16x16x32_bf16(
                        false, af[ms].v, false, bf.v, (short)0, acc2[ms][j],
                        false, false);
                }
            }
        }
        __syncthreads();
    }

    // ---- finalize: weighted atomic scatter into out ----
    #pragma unroll
    for (int ms = 0; ms < 4; ++ms) {
        #pragma unroll
        for (int j = 0; j < 8; ++j) {
            const int dcol = (wid + j * 8) * 16 + l16;
            #pragma unroll
            for (int r = 0; r < 8; ++r) {
                const int tr = ms * 16 + r + 8 * khalf;
                const int t  = TOKS[tr];
                if (t >= 0) {
                    const float w = WGTS[tr];
                    unsafeAtomicAdd(out + (size_t)t * ND + dcol, w * acc2[ms][j][r]);
                }
            }
        }
    }
}

// ---------------- host-side launcher ----------------
extern "C" void kernel_launch(void* const* d_in, const int* in_sizes, int n_in,
                              void* d_out, int out_size, void* d_ws, size_t ws_size,
                              hipStream_t stream)
{
    const float* x     = (const float*)d_in[0];   // (B,S,D)
    const float* emb   = (const float*)d_in[1];   // (E,D)
    const float* W1    = (const float*)d_in[2];   // (E,D,H)
    const float* b1    = (const float*)d_in[3];   // (E,H)
    const float* W2    = (const float*)d_in[4];   // (E,H,D)
    const float* b2    = (const float*)d_in[5];   // (E,D)
    const float* cache = (const float*)d_in[6];   // (E,)
    float* out = (float*)d_out;

    // workspace carve-up
    unsigned short* w1t = (unsigned short*)d_ws;               // (E,H,D) bf16
    unsigned short* w2t = w1t + (size_t)NE * ND * NH;          // (E,D,H) bf16
    float* enorm2 = (float*)(w2t + (size_t)NE * ND * NH);      // (E,)
    int*   eidx   = (int*)(enorm2 + NE);                       // (N,2)
    float* ewgt   = (float*)(eidx + NTOK * 2);                 // (N,2)
    int*   counts = (int*)(ewgt + NTOK * 2);                   // (E,)
    int*   ltok   = counts + NE;                               // (E,CAP)
    float* lwt    = (float*)(ltok + NE * CAP);                 // (E,CAP)

    // 1) bf16 transposed weights (fits MI455X 192MB L2 -> GEMMs hit L2)
    transpose_bf16_kernel<<<dim3(NH / 32, ND / 32, NE), 256, 0, stream>>>(W1, w1t, ND, NH);
    transpose_bf16_kernel<<<dim3(ND / 32, NH / 32, NE), 256, 0, stream>>>(W2, w2t, NH, ND);

    // 2) routing
    enorm_kernel<<<1, 256, 0, stream>>>(emb, enorm2);
    route_kernel<<<(NTOK + 7) / 8, 256, 0, stream>>>(x, emb, enorm2, cache, eidx, ewgt);
    zero_counts_kernel<<<1, 32, 0, stream>>>(counts);
    scatter_kernel<<<(NTOK + 255) / 256, 256, 0, stream>>>(eidx, ewgt, counts, ltok, lwt);

    // 3) out = sum_k w_k * b2[e_k]  (softmax weights sum to 1)
    init_out_kernel<<<(NTOK * ND) / 256, 256, 0, stream>>>(b2, eidx, ewgt, out);

    // 4) fused expert FFN with bf16 WMMA
    const size_t shmem = (size_t)MT * ND * 2 + (size_t)MT * HC * 2
                       + (size_t)MT * 4 + (size_t)MT * 4;      // ~136.5 KB
    ffn_kernel<<<dim3(CAP / MT, NE), 256, shmem, stream>>>(
        x, w1t, w2t, b1, counts, ltok, lwt, out);
}
